// SlotAttention_63471026700838
// MI455X (gfx1250) — matrix operand order repaired and checked by reading, not compiled, not run on hardware
//
#include <hip/hip_runtime.h>
#include <hip/hip_bf16.h>
#include <math.h>

#define BB  32
#define HWX 4096
#define NS  16
#define EE  256
#define FFD 1024

typedef __attribute__((ext_vector_type(16))) _Float16 v16h;
typedef __attribute__((ext_vector_type(8)))  float    v8f;
typedef __attribute__((ext_vector_type(4)))  float    f32x4;

union HVec { v16h v; _Float16 h[16]; f32x4 f4[2]; };

// ---------------------------------------------------------------------------
// WMMA operand loaders (wave32 16-bit; this code's contiguous k order, not the ISA table order)
// A 16x32 (MxK): lanes 0-15 -> M=lane, K 0..15; lanes 16-31 -> K 16..31
// B 32x16 (KxN): lanes 0-15 -> N=lane, K 0..15; lanes 16-31 -> K 16..31 (2 K per VGPR)
// ---------------------------------------------------------------------------
__device__ __forceinline__ v16h load_a_f16(const _Float16* __restrict__ A, int lda,
                                           int kk, int lane) {
  int m    = lane & 15;
  int koff = kk + ((lane < 16) ? 0 : 16);
  HVec r;
  r.f4[0] = *reinterpret_cast<const f32x4*>(A + (size_t)m * lda + koff);
  r.f4[1] = *reinterpret_cast<const f32x4*>(A + (size_t)m * lda + koff + 8);
  return r.v;
}

// NT: B operand comes from row-major weight W[N][K] (K contiguous) -> contiguous load
__device__ __forceinline__ v16h load_b_nt(const _Float16* __restrict__ Wrow, int ldb,
                                          int kk, int lane) {
  int n     = lane & 15;
  int kbase = kk + ((lane < 16) ? 0 : 16);
  const _Float16* p = Wrow + (size_t)n * ldb + kbase;
  HVec r;
  r.f4[0] = *reinterpret_cast<const f32x4*>(p);
  r.f4[1] = *reinterpret_cast<const f32x4*>(p + 8);
  return r.v;
}

// NN: B operand from row-major matrix Bm[K][N] -> strided gather per K
__device__ __forceinline__ v16h load_b_nn(const _Float16* __restrict__ Bm, int ldb,
                                          int kk, int ncol, int lane) {
  int n     = ncol + (lane & 15);
  int kbase = kk + ((lane < 16) ? 0 : 16);
  const _Float16* p = Bm + (size_t)kbase * ldb + n;
  HVec r;
#pragma unroll
  for (int t = 0; t < 16; ++t) r.h[t] = p[(size_t)t * ldb];
  return r.v;
}

// Fused-LayerNorm A loader: reads raw f32 input, applies (x-m)*rstd*g+b, emits f16
__device__ __forceinline__ v16h load_a_ln(const float* __restrict__ X,
                                          const float2* __restrict__ st,
                                          const float* __restrict__ g,
                                          const float* __restrict__ bt,
                                          int row, int kk, int lane) {
  int koff = kk + ((lane < 16) ? 0 : 16);
  float2 s = st[row];
  const float* x = X + (size_t)row * EE;
  HVec r;
#pragma unroll
  for (int t = 0; t < 8; ++t) {
    int c = koff + t;
    r.h[t] = (_Float16)((x[c] - s.x) * s.y * g[c] + bt[c]);
  }
#pragma unroll
  for (int t = 0; t < 8; ++t) {
    int c = koff + 8 + t;
    r.h[8 + t] = (_Float16)((x[c] - s.x) * s.y * g[c] + bt[c]);
  }
  return r.v;
}

__device__ __forceinline__ float gelu_tanh(float x) {
  float x3 = x * x * x;
  return 0.5f * x * (1.0f + tanhf(0.7978845608028654f * (x + 0.044715f * x3)));
}

// ---------------------------------------------------------------------------
// Generic batched GEMM: C = A(MxK) * B^T|B (NT weights / NN matrix), f16 in, f32 acc
// One wave computes a 16x64 output tile, K-step 32 (v_wmma_f32_16x16x32_f16).
// EPI: 0 store f32 | 1 store f16*alpha | 2 store f16 gelu(x+bias) | 3 store f32 x+bias+resid
// ---------------------------------------------------------------------------
template <int EPI, bool BNN>
__global__ void gemm_wmma(const _Float16* __restrict__ A, const _Float16* __restrict__ Bw,
                          void* __restrict__ Cout,
                          const float* __restrict__ bias, const float* __restrict__ resid,
                          int M, int N, int K, int lda, int ldb, int ldc,
                          long batchA, long batchB, long batchC, float alpha) {
  int wave    = (int)((blockIdx.x * blockDim.x + threadIdx.x) >> 5);
  int lane    = threadIdx.x & 31;
  int tiles_n = N >> 6;
  int tiles_m = M >> 4;
  if (wave >= tiles_m * tiles_n) return;  // wave-uniform exit: EXEC stays all-ones
  int tm = wave / tiles_n;
  int tn = wave - tm * tiles_n;
  const _Float16* Ab = A + (long)blockIdx.y * batchA + (long)tm * 16 * lda;
  const _Float16* Bb = Bw + (long)blockIdx.y * batchB;

  v8f acc[4] = {};
  for (int kk = 0; kk < K; kk += 32) {
    v16h a = load_a_f16(Ab, lda, kk, lane);
#pragma unroll
    for (int s = 0; s < 4; ++s) {
      int nb = (tn << 6) + (s << 4);
      v16h b;
      if (BNN) b = load_b_nn(Bb, ldb, kk, nb, lane);
      else     b = load_b_nt(Bb + (size_t)nb * ldb, ldb, kk, lane);
      acc[s] = __builtin_amdgcn_wmma_f32_16x16x32_f16(false, a, false, b,
                                                      (short)0, acc[s], false, false);
    }
  }

  int mbase = (tm << 4) + ((lane < 16) ? 0 : 8);
#pragma unroll
  for (int s = 0; s < 4; ++s) {
    int col = (tn << 6) + (s << 4) + (lane & 15);
#pragma unroll
    for (int r = 0; r < 8; ++r) {
      long idx = (long)blockIdx.y * batchC + (long)(mbase + r) * ldc + col;
      float v = acc[s][r];
      if      (EPI == 0) ((float*)Cout)[idx]    = v;
      else if (EPI == 1) ((_Float16*)Cout)[idx] = (_Float16)(v * alpha);
      else if (EPI == 2) ((_Float16*)Cout)[idx] = (_Float16)gelu_tanh(v + bias[col]);
      else               ((float*)Cout)[idx]    = v + bias[col] + resid[idx];
    }
  }
}

// Dual-output projection: k = LN(x) @ Wk^T, v = LN(x) @ Wv^T, sharing the LN'd A tile
__global__ void kvproj_kernel(const float* __restrict__ X, const float2* __restrict__ st,
                              const float* __restrict__ g, const float* __restrict__ bt,
                              const _Float16* __restrict__ Wk, const _Float16* __restrict__ Wv,
                              _Float16* __restrict__ Ko, _Float16* __restrict__ Vo, int Mrows) {
  int wave = (int)((blockIdx.x * blockDim.x + threadIdx.x) >> 5);
  int lane = threadIdx.x & 31;
  int tiles = (Mrows >> 4) * 4;  // N=256 -> 4 tiles of 64
  if (wave >= tiles) return;
  int tm = wave >> 2, tn = wave & 3;
  int row = (tm << 4) + (lane & 15);

  v8f ak[4] = {}, av[4] = {};
  for (int kk = 0; kk < EE; kk += 32) {
    v16h a = load_a_ln(X, st, g, bt, row, kk, lane);
#pragma unroll
    for (int s = 0; s < 4; ++s) {
      int nb = (tn << 6) + (s << 4);
      v16h bk = load_b_nt(Wk + (size_t)nb * EE, EE, kk, lane);
      v16h bv = load_b_nt(Wv + (size_t)nb * EE, EE, kk, lane);
      ak[s] = __builtin_amdgcn_wmma_f32_16x16x32_f16(false, a, false, bk, (short)0, ak[s], false, false);
      av[s] = __builtin_amdgcn_wmma_f32_16x16x32_f16(false, a, false, bv, (short)0, av[s], false, false);
    }
  }
  int mbase = (tm << 4) + ((lane < 16) ? 0 : 8);
#pragma unroll
  for (int s = 0; s < 4; ++s) {
    int col = (tn << 6) + (s << 4) + (lane & 15);
#pragma unroll
    for (int r = 0; r < 8; ++r) {
      size_t idx = (size_t)(mbase + r) * EE + col;
      Ko[idx] = (_Float16)ak[s][r];
      Vo[idx] = (_Float16)av[s][r];
    }
  }
}

// Per-row mean/rstd over 256 columns (one wave per row)
__global__ void ln_stats_kernel(const float* __restrict__ X, float2* __restrict__ st, int rows) {
  int lane = threadIdx.x & 31;
  int row  = blockIdx.x * (blockDim.x >> 5) + (threadIdx.x >> 5);
  if (row >= rows) return;
  const float* x = X + (size_t)row * EE;
  float s = 0.f, s2 = 0.f;
#pragma unroll
  for (int t = 0; t < 8; ++t) { float v = x[lane * 8 + t]; s += v; s2 += v * v; }
  for (int off = 16; off; off >>= 1) { s += __shfl_xor(s, off, 32); s2 += __shfl_xor(s2, off, 32); }
  if (lane == 0) {
    float m = s * (1.0f / EE);
    st[row] = make_float2(m, rsqrtf(s2 * (1.0f / EE) - m * m + 1e-5f));
  }
}

// LayerNorm rows of f32 -> f16 (one wave per 256-col row)
__global__ void ln_rows_f16_kernel(const float* __restrict__ X, const float* __restrict__ g,
                                   const float* __restrict__ bt, _Float16* __restrict__ out,
                                   int rows) {
  int lane = threadIdx.x & 31;
  int row  = blockIdx.x * (blockDim.x >> 5) + (threadIdx.x >> 5);
  if (row >= rows) return;
  const float* x = X + (size_t)row * EE;
  float s = 0.f, s2 = 0.f;
#pragma unroll
  for (int t = 0; t < 8; ++t) { float v = x[lane * 8 + t]; s += v; s2 += v * v; }
  for (int off = 16; off; off >>= 1) { s += __shfl_xor(s, off, 32); s2 += __shfl_xor(s2, off, 32); }
  float m = s * (1.0f / EE);
  float rstd = rsqrtf(s2 * (1.0f / EE) - m * m + 1e-5f);
#pragma unroll
  for (int t = 0; t < 8; ++t) {
    int c = lane * 8 + t;
    out[(size_t)row * EE + c] = (_Float16)((x[c] - m) * rstd * g[c] + bt[c]);
  }
}

__global__ void cvt_f16_kernel(const float* __restrict__ in, _Float16* __restrict__ out, int n) {
  int i = blockIdx.x * blockDim.x + threadIdx.x;
  if (i < n) out[i] = (_Float16)in[i];
}

// Softmax across the 16-slot axis (axis=1) for each (b, hw) column; coalesced per q row.
__global__ void softmax_col_kernel(const float* __restrict__ logit, float* __restrict__ attn) {
  long idx = (long)blockIdx.x * blockDim.x + threadIdx.x;
  if (idx >= (long)BB * HWX) return;
  long b = idx / HWX, hw = idx - b * HWX;
  const float* p = logit + b * (long)NS * HWX + hw;
  float v[NS], mx = -3.402823466e38f;
#pragma unroll
  for (int q = 0; q < NS; ++q) { v[q] = p[(long)q * HWX]; mx = fmaxf(mx, v[q]); }
  float s = 0.f;
#pragma unroll
  for (int q = 0; q < NS; ++q) { v[q] = expf(v[q] - mx); s += v[q]; }
  float inv = 1.0f / s;
  float* o = attn + b * (long)NS * HWX + hw;
#pragma unroll
  for (int q = 0; q < NS; ++q) o[(long)q * HWX] = v[q] * inv;
}

// Deterministic row sum over HW per (b,q)
__global__ void row_sum_kernel(const float* __restrict__ attn, float* __restrict__ rsum) {
  __shared__ float sm[256];
  const float* p = attn + (long)blockIdx.x * HWX;
  float s = 0.f;
  for (int i = threadIdx.x; i < HWX; i += 256) s += p[i];
  sm[threadIdx.x] = s;
  __syncthreads();
  for (int off = 128; off; off >>= 1) {
    if ((int)threadIdx.x < off) sm[threadIdx.x] += sm[threadIdx.x + off];
    __syncthreads();
  }
  if (threadIdx.x == 0) rsum[blockIdx.x] = sm[0];
}

__global__ void attnw_kernel(const float* __restrict__ attn, const float* __restrict__ rsum,
                             _Float16* __restrict__ out) {
  long i = (long)blockIdx.x * blockDim.x + threadIdx.x;
  if (i >= (long)BB * NS * HWX) return;
  out[i] = (_Float16)(attn[i] / (rsum[i >> 12] + 1e-5f));
}

// GRU gates (PyTorch GRUCell semantics, gate order r,z,n)
__global__ void gru_gate_kernel(const float* __restrict__ gi, const float* __restrict__ gh,
                                const float* __restrict__ b_ih, const float* __restrict__ b_hh,
                                const float* __restrict__ hprev, float* __restrict__ out) {
  int i = blockIdx.x * blockDim.x + threadIdx.x;
  if (i >= 512 * EE) return;
  int row = i >> 8, c = i & 255;
  const float* gir = gi + (size_t)row * 3 * EE;
  const float* ghr = gh + (size_t)row * 3 * EE;
  float ir = gir[c] + b_ih[c],            hr = ghr[c] + b_hh[c];
  float iz = gir[EE + c] + b_ih[EE + c],  hz = ghr[EE + c] + b_hh[EE + c];
  float in = gir[2*EE + c] + b_ih[2*EE + c], hn = ghr[2*EE + c] + b_hh[2*EE + c];
  float r = 1.0f / (1.0f + expf(-(ir + hr)));
  float z = 1.0f / (1.0f + expf(-(iz + hz)));
  float n = tanhf(in + r * hn);
  out[i] = (1.0f - z) * n + z * hprev[i];
}

// ---------------------------------------------------------------------------
static inline dim3 gemm_grid(int M, int N, int batches) {
  int waves = (M >> 4) * (N >> 6);
  int gx = (waves * 32 + 255) / 256;
  return dim3((unsigned)gx, (unsigned)batches, 1);
}

extern "C" void kernel_launch(void* const* d_in, const int* in_sizes, int n_in,
                              void* d_out, int out_size, void* d_ws, size_t ws_size,
                              hipStream_t stream) {
  (void)in_sizes; (void)n_in; (void)out_size; (void)ws_size;
  const float* input = (const float*)d_in[0];
  const float* query = (const float*)d_in[1];
  const float* g_kv  = (const float*)d_in[2];
  const float* b_kv  = (const float*)d_in[3];
  const float* w_k   = (const float*)d_in[4];
  const float* w_v   = (const float*)d_in[5];
  const float* g_q   = (const float*)d_in[6];
  const float* b_q   = (const float*)d_in[7];
  const float* w_q   = (const float*)d_in[8];
  const float* w_ih  = (const float*)d_in[9];
  const float* w_hh  = (const float*)d_in[10];
  const float* b_ih  = (const float*)d_in[11];
  const float* b_hh  = (const float*)d_in[12];
  const float* g_2   = (const float*)d_in[13];
  const float* b_2   = (const float*)d_in[14];
  const float* w_f1  = (const float*)d_in[15];
  const float* b_f1  = (const float*)d_in[16];
  const float* w_f2  = (const float*)d_in[17];
  const float* b_f2  = (const float*)d_in[18];

  const int Mkv = BB * HWX;  // 131072 rows
  const int MQ  = BB * NS;   // 512 rows

  char* ws = (char*)d_ws;
  size_t off = 0;
  auto alloc = [&](size_t bytes) -> char* {
    char* p = ws + off;
    off += (bytes + 255) & ~(size_t)255;
    return p;
  };
  float2*   stats = (float2*)  alloc(sizeof(float2) * Mkv);
  _Float16* K16   = (_Float16*)alloc(2ull * Mkv * EE);      // 64 MB (k+v f16 fit in L2)
  _Float16* V16   = (_Float16*)alloc(2ull * Mkv * EE);      // 64 MB
  _Float16* wk16  = (_Float16*)alloc(2ull * EE * EE);
  _Float16* wv16  = (_Float16*)alloc(2ull * EE * EE);
  _Float16* wq16  = (_Float16*)alloc(2ull * EE * EE);
  _Float16* wih16 = (_Float16*)alloc(2ull * 3 * EE * EE);
  _Float16* whh16 = (_Float16*)alloc(2ull * 3 * EE * EE);
  _Float16* wf116 = (_Float16*)alloc(2ull * FFD * EE);
  _Float16* wf216 = (_Float16*)alloc(2ull * EE * FFD);
  _Float16* qn16  = (_Float16*)alloc(2ull * MQ * EE);
  _Float16* qp16  = (_Float16*)alloc(2ull * MQ * EE);
  float*    logit = (float*)   alloc(4ull * BB * NS * HWX); // 8 MB
  float*    rsum  = (float*)   alloc(4ull * MQ);
  _Float16* aw16  = (_Float16*)alloc(2ull * BB * NS * HWX); // 4 MB
  _Float16* upd16 = (_Float16*)alloc(2ull * MQ * EE);
  _Float16* q16   = (_Float16*)alloc(2ull * MQ * EE);
  float*    gi    = (float*)   alloc(4ull * MQ * 3 * EE);
  float*    gh    = (float*)   alloc(4ull * MQ * 3 * EE);
  float*    slots = (float*)   alloc(4ull * MQ * EE);
  _Float16* ln216 = (_Float16*)alloc(2ull * MQ * EE);
  _Float16* h116  = (_Float16*)alloc(2ull * MQ * FFD);
  float*    qbufA = (float*)   alloc(4ull * MQ * EE);
  float*    qbufB = (float*)   alloc(4ull * MQ * EE);

  float* qOut    = (float*)d_out;            // (B,N,E)
  float* attnOut = (float*)d_out + MQ * EE;  // (B,N,HW) — last iteration persists

  // f32 -> f16 weight conversions
  auto cvt = [&](const float* src, _Float16* dst, int n) {
    cvt_f16_kernel<<<(n + 255) / 256, 256, 0, stream>>>(src, dst, n);
  };
  cvt(w_k,  wk16,  EE * EE);
  cvt(w_v,  wv16,  EE * EE);
  cvt(w_q,  wq16,  EE * EE);
  cvt(w_ih, wih16, 3 * EE * EE);
  cvt(w_hh, whh16, 3 * EE * EE);
  cvt(w_f1, wf116, FFD * EE);
  cvt(w_f2, wf216, EE * FFD);

  // kv = LN(input); k = kv@Wk^T, v = kv@Wv^T  (LN fused into the A-tile loader)
  ln_stats_kernel<<<Mkv / 8, 256, 0, stream>>>(input, stats, Mkv);
  kvproj_kernel<<<((Mkv / 16) * 4 * 32) / 256, 256, 0, stream>>>(
      input, stats, g_kv, b_kv, wk16, wv16, K16, V16, Mkv);

  const float scale = 0.0625f;  // E^-0.5
  for (int it = 0; it < 3; ++it) {
    const float* qin = (it == 0) ? query : (it == 1 ? qbufA : qbufB);
    float* qout      = (it == 2) ? qOut  : (it == 0 ? qbufA : qbufB);

    // qn = LN(q); qp = (qn @ Wq^T) * scale  -> f16
    ln_rows_f16_kernel<<<MQ / 8, 256, 0, stream>>>(qin, g_q, b_q, qn16, MQ);
    gemm_wmma<1, false><<<gemm_grid(MQ, EE, 1), 256, 0, stream>>>(
        qn16, wq16, qp16, nullptr, nullptr, MQ, EE, EE, EE, EE, EE, 0, 0, 0, scale);

    // logit[b] = qp[b] @ k[b]^T  (batched NT, M=16, N=4096, K=256)
    gemm_wmma<0, false><<<gemm_grid(NS, HWX, BB), 256, 0, stream>>>(
        qp16, K16, logit, nullptr, nullptr, NS, HWX, EE, EE, EE, HWX,
        (long)NS * EE, (long)HWX * EE, (long)NS * HWX, 1.0f);

    // softmax over slot axis -> attn (written into d_out region), renormalize over HW
    softmax_col_kernel<<<(BB * HWX) / 256, 256, 0, stream>>>(logit, attnOut);
    row_sum_kernel<<<BB * NS, 256, 0, stream>>>(attnOut, rsum);
    attnw_kernel<<<(BB * NS * HWX) / 256, 256, 0, stream>>>(attnOut, rsum, aw16);

    // updates[b] = attn_w[b] @ v[b]  (batched NN, M=16, N=256, K=4096)
    gemm_wmma<1, true><<<gemm_grid(NS, EE, BB), 256, 0, stream>>>(
        aw16, V16, upd16, nullptr, nullptr, NS, EE, HWX, HWX, EE, EE,
        (long)NS * HWX, (long)HWX * EE, (long)NS * EE, 1.0f);

    // GRU: gi = updates @ Wih^T, gh = h @ Whh^T, then gates
    cvt_f16_kernel<<<(MQ * EE) / 256, 256, 0, stream>>>(qin, q16, MQ * EE);
    gemm_wmma<0, false><<<gemm_grid(MQ, 3 * EE, 1), 256, 0, stream>>>(
        upd16, wih16, gi, nullptr, nullptr, MQ, 3 * EE, EE, EE, EE, 3 * EE, 0, 0, 0, 1.0f);
    gemm_wmma<0, false><<<gemm_grid(MQ, 3 * EE, 1), 256, 0, stream>>>(
        q16, whh16, gh, nullptr, nullptr, MQ, 3 * EE, EE, EE, EE, 3 * EE, 0, 0, 0, 1.0f);
    gru_gate_kernel<<<(MQ * EE) / 256, 256, 0, stream>>>(gi, gh, b_ih, b_hh, qin, slots);

    // FFN: q = slots + W2 @ gelu(W1 @ LN(slots) + b1) + b2 (bias/gelu/residual fused)
    ln_rows_f16_kernel<<<MQ / 8, 256, 0, stream>>>(slots, g_2, b_2, ln216, MQ);
    gemm_wmma<2, false><<<gemm_grid(MQ, FFD, 1), 256, 0, stream>>>(
        ln216, wf116, h116, b_f1, nullptr, MQ, FFD, EE, EE, EE, FFD, 0, 0, 0, 1.0f);
    gemm_wmma<3, false><<<gemm_grid(MQ, EE, 1), 256, 0, stream>>>(
        h116, wf216, qout, b_f2, slots, MQ, EE, FFD, FFD, FFD, EE, 0, 0, 0, 1.0f);
  }
}
